// SISVAE_78056735637993
// MI455X (gfx1250) — matrix operand order repaired and checked by reading, not compiled, not run on hardware
//
#include <hip/hip_runtime.h>
#include <hip/hip_bf16.h>

// ---------------------------------------------------------------------------
// SISVAE forward (B=64, T=128, X=64, H=512) as a persistent wave32 kernel.
// GEMMs use v_wmma_f32_16x16x32_f16 with SWAPPED operand roles:
//   D(16n x 16m) = Wt_tile(16n x 32k) x act^T_tile(32k x 16m)
// so each lane's 8 accumulator values are CONTIGUOUS in n -> vectorized
// (b128) epilogue loads/stores. K-reduction uses 4 independent accumulators
// to eliminate WMMA->WMMA RAW hazard NOPs (5-cycle bubble per dependent pair).
// Sequential recurrence handled with device-wide phase barriers.
// ---------------------------------------------------------------------------

#define Bd 64
#define Td 128
#define Xd 64
#define Hd 512

#define NBLK 16
#define TPB 256
#define NWAVE (NBLK * TPB / 32)   // 128 waves
#define NTHR  (NBLK * TPB)        // 4096 threads

typedef __attribute__((ext_vector_type(16))) _Float16 v16h;
typedef __attribute__((ext_vector_type(8)))  _Float16 v8h;
typedef __attribute__((ext_vector_type(8)))  float    v8f;

static_assert(NTHR * 8 == Bd * Hd, "elementwise stages assume 8 elems/thread");

// ---------------- workspace layout (bytes) ----------------
constexpr size_t O_WT_X2Z   = 0;                              // [512][64]  f16
constexpr size_t O_WT_XH2Z  = O_WT_X2Z   + (size_t)512*64*2;  // [1024][512]
constexpr size_t O_WT_Z2X   = O_WT_XH2Z  + (size_t)1024*512*2;// [512][512]
constexpr size_t O_WT_ZH2X  = O_WT_Z2X   + (size_t)512*512*2; // [128][512]
constexpr size_t O_WT_DEC   = O_WT_ZH2X  + (size_t)128*512*2; // [512][512]
constexpr size_t O_WT_PRIOR = O_WT_DEC   + (size_t)512*512*2; // [1024][512]
constexpr size_t O_WT_IH    = O_WT_PRIOR + (size_t)1024*512*2;// [1536][512]
constexpr size_t O_WT_HH    = O_WT_IH    + (size_t)1536*512*2;// [1536][512]
constexpr size_t O_X16      = O_WT_HH    + (size_t)1536*512*2;// [T][B][X] f16
constexpr size_t O_H16      = O_X16      + (size_t)Td*Bd*Xd*2;
constexpr size_t O_PRE16    = O_H16      + (size_t)Bd*Hd*2;
constexpr size_t O_Z16      = O_PRE16    + (size_t)Bd*Hd*2;
constexpr size_t O_T2_16    = O_Z16      + (size_t)Bd*Hd*2;
constexpr size_t O_DEC16    = O_T2_16    + (size_t)Bd*Hd*2;
constexpr size_t O_H32      = O_DEC16    + (size_t)Bd*Hd*2;   // f32
constexpr size_t O_T1       = O_H32      + (size_t)Bd*Hd*4;   // [B][1024]
constexpr size_t O_PRIOR    = O_T1       + (size_t)Bd*1024*4; // [B][1024]
constexpr size_t O_GH       = O_PRIOR    + (size_t)Bd*1024*4; // [B][1536]
constexpr size_t O_GI       = O_GH       + (size_t)Bd*1536*4; // [B][1536]
constexpr size_t O_MUXT     = O_GI       + (size_t)Bd*1536*4; // [2][B][X]
constexpr size_t O_THXT     = O_MUXT     + (size_t)2*Bd*Xd*4; // [2][B][X]
constexpr size_t O_PART     = O_THXT     + (size_t)2*Bd*Xd*4; // 3*NBLK f32 (padded)
constexpr size_t O_CNT      = O_PART     + 256;               // barrier counter
constexpr size_t WS_NEEDED  = O_CNT + 256;

// ---------------- device helpers ----------------
__device__ inline float softplus_f(float x) {
  return (x > 20.f) ? x : log1pf(expf(x));
}
__device__ inline float sigmoid_f(float x) {
  return 1.f / (1.f + expf(-x));
}
__device__ inline float kld_term(float m1, float s1, float m2, float s2) {
  float d = m1 - m2;
  return 0.5f * (2.f * (logf(s2) - logf(s1)) + (s1 * s1 + d * d) / (s2 * s2) - 1.f);
}

// 16x16 tile of (act @ W) computed as D = Wt_tile (A op) x act^T_tile (B op).
//   act : [M][K] row-major f16  (M = batch rows)
//   Wt  : [N][K] row-major f16  (W pre-transposed)
// A operand in contiguous k = 16*lh + s order (same as B): lane row = n0+l15, half lh -> two 16B loads.
// B operand in contiguous k = 16*lh + s order (same as A): lane col = m0+l15, kh=lh -> one 32B load.
// D layout: lane holds m = m0+l15 fixed, n = n0 + lh*8 + q (q=0..7) CONTIGUOUS.
// 4 round-robin accumulators break the WMMA->WMMA RAW dependency chain.
template <int K>
__device__ inline v8f wmma_tile(const _Float16* act, int lda,
                                const _Float16* Wt, int ldw,
                                int m0, int n0) {
  const int lane = threadIdx.x & 31;
  const int l15  = lane & 15;
  const int lh   = lane >> 4;
  const _Float16* pa = Wt  + (size_t)(n0 + l15) * ldw + lh * 16;  // A: weights
  const _Float16* pb = act + (size_t)(m0 + l15) * lda + lh * 16;  // B: activations
  v8f acc0 = {}, acc1 = {}, acc2 = {}, acc3 = {};
#pragma unroll
  for (int c = 0; c < K / 32; ++c) {
    const int k = c * 32;
    v8h alo = *(const v8h*)(pa + k);
    v8h ahi = *(const v8h*)(pa + k + 8);
    v16h a  = __builtin_shufflevector(alo, ahi, 0,1,2,3,4,5,6,7,8,9,10,11,12,13,14,15);
    v16h b  = *(const v16h*)(pb + k);
    switch (c & 3) {
      case 0:
        acc0 = __builtin_amdgcn_wmma_f32_16x16x32_f16(false, a, false, b,
                                                      (short)0, acc0, false, false);
        break;
      case 1:
        acc1 = __builtin_amdgcn_wmma_f32_16x16x32_f16(false, a, false, b,
                                                      (short)0, acc1, false, false);
        break;
      case 2:
        acc2 = __builtin_amdgcn_wmma_f32_16x16x32_f16(false, a, false, b,
                                                      (short)0, acc2, false, false);
        break;
      default:
        acc3 = __builtin_amdgcn_wmma_f32_16x16x32_f16(false, a, false, b,
                                                      (short)0, acc3, false, false);
        break;
    }
  }
  return (acc0 + acc1) + (acc2 + acc3);
}

__device__ inline void grid_barrier(unsigned* cnt, unsigned seq) {
  __syncthreads();
  if (threadIdx.x == 0) {
    __threadfence();
    atomicAdd(cnt, 1u);
    const unsigned target = seq * (unsigned)NBLK;
    while (atomicAdd(cnt, 0u) < target) { __builtin_amdgcn_s_sleep(2); }
    __threadfence();
  }
  __syncthreads();
}

__device__ inline float block_reduce(float v, float* red) {
  red[threadIdx.x] = v;
  __syncthreads();
  for (int s = TPB / 2; s > 0; s >>= 1) {
    if (threadIdx.x < (unsigned)s) red[threadIdx.x] += red[threadIdx.x + s];
    __syncthreads();
  }
  float r = red[0];
  __syncthreads();
  return r;
}

// ---------------- prep kernel: weight transpose fp32->f16, x relayout, init ----
__device__ inline void transpose_to_f16(const float* W, _Float16* Wt,
                                        int K, int N, int tid, int nt) {
  const int total = K * N;
  for (int o = tid; o < total; o += nt) {
    int n = o / K, k = o - n * K;
    Wt[o] = (_Float16)W[(size_t)k * N + n];
  }
}

__global__ void sisvae_prep(const float* Wx2z, const float* Wxh2z, const float* Wz2x,
                            const float* Wzh2x, const float* Wdec, const float* Wprior,
                            const float* Wih, const float* Whh, const float* x,
                            char* ws) {
  const int tid = blockIdx.x * blockDim.x + threadIdx.x;
  const int nt  = gridDim.x * blockDim.x;
  transpose_to_f16(Wx2z,  (_Float16*)(ws + O_WT_X2Z),   64, 512,  tid, nt);
  transpose_to_f16(Wxh2z, (_Float16*)(ws + O_WT_XH2Z), 512, 1024, tid, nt);
  transpose_to_f16(Wz2x,  (_Float16*)(ws + O_WT_Z2X),  512, 512,  tid, nt);
  transpose_to_f16(Wzh2x, (_Float16*)(ws + O_WT_ZH2X), 512, 128,  tid, nt);
  transpose_to_f16(Wdec,  (_Float16*)(ws + O_WT_DEC),  512, 512,  tid, nt);
  transpose_to_f16(Wprior,(_Float16*)(ws + O_WT_PRIOR),512, 1024, tid, nt);
  transpose_to_f16(Wih,   (_Float16*)(ws + O_WT_IH),   512, 1536, tid, nt);
  transpose_to_f16(Whh,   (_Float16*)(ws + O_WT_HH),   512, 1536, tid, nt);
  // x: [B][T][X] fp32 -> [T][B][X] f16
  _Float16* x16 = (_Float16*)(ws + O_X16);
  for (int o = tid; o < Td * Bd * Xd; o += nt) {
    int t = o >> 12, r = o & 4095, b = r >> 6, xx = r & 63;
    x16[o] = (_Float16)x[((size_t)b * Td + t) * Xd + xx];
  }
  // h0 = 0
  float*    h32 = (float*)(ws + O_H32);
  _Float16* h16 = (_Float16*)(ws + O_H16);
  for (int o = tid; o < Bd * Hd; o += nt) { h32[o] = 0.f; h16[o] = (_Float16)0.f; }
  if (tid == 0) *(unsigned*)(ws + O_CNT) = 0u;
}

// ---------------- persistent recurrence kernel ----------------
__global__ __launch_bounds__(TPB) void sisvae_persistent(
    const float* __restrict__ x, const float* __restrict__ eps_z,
    const float* __restrict__ eps_x,
    const float* __restrict__ b_x2z, const float* __restrict__ b_xh2z,
    const float* __restrict__ b_z2x, const float* __restrict__ b_zh2x,
    const float* __restrict__ b_dec, const float* __restrict__ b_prior,
    const float* __restrict__ b_ih,  const float* __restrict__ b_hh,
    char* ws, float* __restrict__ out) {
  const _Float16* wt_x2z   = (const _Float16*)(ws + O_WT_X2Z);
  const _Float16* wt_xh2z  = (const _Float16*)(ws + O_WT_XH2Z);
  const _Float16* wt_z2x   = (const _Float16*)(ws + O_WT_Z2X);
  const _Float16* wt_zh2x  = (const _Float16*)(ws + O_WT_ZH2X);
  const _Float16* wt_dec   = (const _Float16*)(ws + O_WT_DEC);
  const _Float16* wt_prior = (const _Float16*)(ws + O_WT_PRIOR);
  const _Float16* wt_ih    = (const _Float16*)(ws + O_WT_IH);
  const _Float16* wt_hh    = (const _Float16*)(ws + O_WT_HH);
  const _Float16* x16      = (const _Float16*)(ws + O_X16);
  _Float16* h16   = (_Float16*)(ws + O_H16);
  _Float16* pre16 = (_Float16*)(ws + O_PRE16);
  _Float16* z16   = (_Float16*)(ws + O_Z16);
  _Float16* t2_16 = (_Float16*)(ws + O_T2_16);
  _Float16* dec16 = (_Float16*)(ws + O_DEC16);
  float* h32   = (float*)(ws + O_H32);
  float* t1    = (float*)(ws + O_T1);
  float* prior = (float*)(ws + O_PRIOR);
  float* gh    = (float*)(ws + O_GH);
  float* gi    = (float*)(ws + O_GI);
  float* muxt  = (float*)(ws + O_MUXT);
  float* thxt  = (float*)(ws + O_THXT);
  float* part  = (float*)(ws + O_PART);
  unsigned* cnt = (unsigned*)(ws + O_CNT);

  const int lane = threadIdx.x & 31;
  const int l15  = lane & 15;
  const int lh   = lane >> 4;
  const int wave = blockIdx.x * (TPB / 32) + (threadIdx.x >> 5);
  const int tid  = blockIdx.x * TPB + threadIdx.x;
  unsigned bseq = 0;

  __shared__ float red[TPB];

  float mse_local = 0.f, zkld_local = 0.f, skld_local = 0.f;

  for (int t = 0; t < Td; ++t) {
    const _Float16* xt16 = x16 + (size_t)t * (Bd * Xd);
    float* muc = muxt + (t & 1) * (Bd * Xd);
    float* thc = thxt + (t & 1) * (Bd * Xd);

    // ---- Stage A: pre = x_t @ W_x2z + b_x2z + h    [64 x 512]
    for (int j = wave; j < 4 * 32; j += NWAVE) {
      int mt = j >> 5, nt = j & 31;
      v8f a = wmma_tile<64>(xt16, Xd, wt_x2z, Xd, mt * 16, nt * 16);
      const int m = mt * 16 + l15;
      const int nb = nt * 16 + lh * 8;            // 8 contiguous n per lane
      v8f bv = *(const v8f*)(b_x2z + nb);
      v8f hv = *(const v8f*)(h32 + (size_t)m * Hd + nb);
      v8h o;
#pragma unroll
      for (int q = 0; q < 8; ++q) o[q] = (_Float16)(a[q] + bv[q] + hv[q]);
      *(v8h*)(pre16 + (size_t)m * Hd + nb) = o;
    }
    grid_barrier(cnt, ++bseq);

    // ---- Stage B: t1 = pre @ W_xh2z + b_xh2z       [64 x 1024]
    for (int j = wave; j < 4 * 64; j += NWAVE) {
      int mt = j >> 6, nt = j & 63;
      v8f a = wmma_tile<512>(pre16, Hd, wt_xh2z, Hd, mt * 16, nt * 16);
      const int m = mt * 16 + l15;
      const int nb = nt * 16 + lh * 8;
      v8f bv = *(const v8f*)(b_xh2z + nb);
      *(v8f*)(t1 + (size_t)m * 1024 + nb) = a + bv;
    }
    grid_barrier(cnt, ++bseq);

    // ---- Stage C: z = eps_z * softplus(th_zt) + mu_zt   (8 contiguous/thread)
    {
      const int i0 = tid * 8;
      const int b = i0 >> 9, n = i0 & 511;
      const float* t1r = t1 + b * 1024;
      v8f mu = *(const v8f*)(t1r + n);
      v8f th = *(const v8f*)(t1r + 512 + n);
      v8f ez = *(const v8f*)(eps_z + ((size_t)b * Td + t) * Hd + n);
      v8h zv;
#pragma unroll
      for (int q = 0; q < 8; ++q)
        zv[q] = (_Float16)(ez[q] * softplus_f(th[q]) + mu[q]);
      *(v8h*)(z16 + i0) = zv;
    }
    grid_barrier(cnt, ++bseq);

    // ---- Stage D: t2 = z@W_z2x+b+h ; dec_in = z@W_dec+b ;
    //               prior = h@W_prior+b ; gh = h@W_hh+b
    for (int j = wave; j < 896; j += NWAVE) {
      if (j < 128) {                       // t2 [64x512]
        int mt = j >> 5, nt = j & 31;
        v8f a = wmma_tile<512>(z16, Hd, wt_z2x, Hd, mt * 16, nt * 16);
        const int m = mt * 16 + l15, nb = nt * 16 + lh * 8;
        v8f bv = *(const v8f*)(b_z2x + nb);
        v8f hv = *(const v8f*)(h32 + (size_t)m * Hd + nb);
        v8h o;
#pragma unroll
        for (int q = 0; q < 8; ++q) o[q] = (_Float16)(a[q] + bv[q] + hv[q]);
        *(v8h*)(t2_16 + (size_t)m * Hd + nb) = o;
      } else if (j < 256) {                // dec_in [64x512]
        int jj = j - 128, mt = jj >> 5, nt = jj & 31;
        v8f a = wmma_tile<512>(z16, Hd, wt_dec, Hd, mt * 16, nt * 16);
        const int m = mt * 16 + l15, nb = nt * 16 + lh * 8;
        v8f bv = *(const v8f*)(b_dec + nb);
        v8h o;
#pragma unroll
        for (int q = 0; q < 8; ++q) o[q] = (_Float16)(a[q] + bv[q]);
        *(v8h*)(dec16 + (size_t)m * Hd + nb) = o;
      } else if (j < 512) {                // prior [64x1024]
        int jj = j - 256, mt = jj >> 6, nt = jj & 63;
        v8f a = wmma_tile<512>(h16, Hd, wt_prior, Hd, mt * 16, nt * 16);
        const int m = mt * 16 + l15, nb = nt * 16 + lh * 8;
        v8f bv = *(const v8f*)(b_prior + nb);
        *(v8f*)(prior + (size_t)m * 1024 + nb) = a + bv;
      } else {                             // gh [64x1536]
        int jj = j - 512, mt = jj / 96, nt = jj % 96;
        v8f a = wmma_tile<512>(h16, Hd, wt_hh, Hd, mt * 16, nt * 16);
        const int m = mt * 16 + l15, nb = nt * 16 + lh * 8;
        v8f bv = *(const v8f*)(b_hh + nb);
        *(v8f*)(gh + (size_t)m * 1536 + nb) = a + bv;
      }
    }
    grid_barrier(cnt, ++bseq);

    // ---- Stage E: xdec = t2@W_zh2x+b -> (mu_xt, softplus th_xt);
    //               gi = dec_in@W_ih+b ; plus z-KLD accumulation
    for (int j = wave; j < 416; j += NWAVE) {
      if (j < 32) {                        // xdec [64x128]
        int mt = j >> 3, nt = j & 7;
        v8f a = wmma_tile<512>(t2_16, Hd, wt_zh2x, Hd, mt * 16, nt * 16);
        const int m = mt * 16 + l15, nb = nt * 16 + lh * 8;
        v8f bv = *(const v8f*)(b_zh2x + nb);
        v8f v = a + bv;
        if (nt < 4) {                      // entire tile is mu_xt
          *(v8f*)(muc + m * Xd + nb) = v;
        } else {                           // entire tile is th_xt
          v8f sp;
#pragma unroll
          for (int q = 0; q < 8; ++q) sp[q] = softplus_f(v[q]);
          *(v8f*)(thc + m * Xd + (nb - Xd)) = sp;
        }
      } else {                             // gi [64x1536]
        int jj = j - 32, mt = jj / 96, nt = jj % 96;
        v8f a = wmma_tile<512>(dec16, Hd, wt_ih, Hd, mt * 16, nt * 16);
        const int m = mt * 16 + l15, nb = nt * 16 + lh * 8;
        v8f bv = *(const v8f*)(b_ih + nb);
        *(v8f*)(gi + (size_t)m * 1536 + nb) = a + bv;
      }
    }
    // z-KLD( prior || posterior ): m1,s1 = prior ; m2,s2 = posterior
    {
      const int i0 = tid * 8;
      const int b = i0 >> 9, n = i0 & 511;
      v8f m1 = *(const v8f*)(prior + b * 1024 + n);
      v8f s1 = *(const v8f*)(prior + b * 1024 + 512 + n);
      v8f m2 = *(const v8f*)(t1 + b * 1024 + n);
      v8f s2 = *(const v8f*)(t1 + b * 1024 + 512 + n);
#pragma unroll
      for (int q = 0; q < 8; ++q)
        zkld_local += kld_term(m1[q], softplus_f(s1[q]), m2[q], softplus_f(s2[q]));
    }
    grid_barrier(cnt, ++bseq);

    // ---- Stage F: GRU update + rec/mse + smooth-KLD
    {
      const int i0 = tid * 8;
      const int b = i0 >> 9, n = i0 & 511;
      const float* gir = gi + b * 1536;
      const float* ghr = gh + b * 1536;
      v8f ir = *(const v8f*)(gir + n);
      v8f iz = *(const v8f*)(gir + 512 + n);
      v8f ig = *(const v8f*)(gir + 1024 + n);
      v8f hr = *(const v8f*)(ghr + n);
      v8f hz = *(const v8f*)(ghr + 512 + n);
      v8f hg = *(const v8f*)(ghr + 1024 + n);
      v8f hv = *(const v8f*)(h32 + i0);
      v8f hnew;
      v8h hnew16;
#pragma unroll
      for (int q = 0; q < 8; ++q) {
        float r  = sigmoid_f(ir[q] + hr[q]);
        float zg = sigmoid_f(iz[q] + hz[q]);
        float nn = tanhf(ig[q] + r * hg[q]);
        float hh = (1.f - zg) * nn + zg * hv[q];
        hnew[q] = hh;
        hnew16[q] = (_Float16)hh;
      }
      *(v8f*)(h32 + i0) = hnew;
      *(v8h*)(h16 + i0) = hnew16;
    }
    for (int i = tid; i < Bd * Xd; i += NTHR) {
      int b = i >> 6, xx = i & 63;
      float mu = muc[i], th = thc[i];
      float rec = eps_x[((size_t)b * Td + t) * Xd + xx] * th + mu;
      out[((size_t)b * Td + t) * Xd + xx] = rec;
      float d = rec - x[((size_t)b * Td + t) * Xd + xx];
      mse_local += d * d;
      if (t >= 1) {
        const float* mup = muxt + ((t & 1) ^ 1) * (Bd * Xd);
        const float* thp = thxt + ((t & 1) ^ 1) * (Bd * Xd);
        skld_local += kld_term(mup[i], thp[i], mu, th);
      }
    }
    grid_barrier(cnt, ++bseq);
  }

  // ---- deterministic final reduction of the three loss sums
  float s;
  s = block_reduce(mse_local, red);
  if (threadIdx.x == 0) part[blockIdx.x] = s;
  s = block_reduce(zkld_local, red);
  if (threadIdx.x == 0) part[NBLK + blockIdx.x] = s;
  s = block_reduce(skld_local, red);
  if (threadIdx.x == 0) part[2 * NBLK + blockIdx.x] = s;
  grid_barrier(cnt, ++bseq);
  if (blockIdx.x == 0 && threadIdx.x == 0) {
    float s0 = 0.f, s1 = 0.f, s2 = 0.f;
    for (int i = 0; i < NBLK; ++i) {
      s0 += part[i]; s1 += part[NBLK + i]; s2 += part[2 * NBLK + i];
    }
    const size_t base = (size_t)Bd * Td * Xd;
    out[base + 0] = s0 / (float)(Bd * Td * Xd);
    out[base + 1] = s1;
    out[base + 2] = s2;
  }
}

// ---------------- host entry ----------------
extern "C" void kernel_launch(void* const* d_in, const int* in_sizes, int n_in,
                              void* d_out, int out_size, void* d_ws, size_t ws_size,
                              hipStream_t stream) {
  (void)in_sizes; (void)n_in; (void)out_size;
  if (ws_size < WS_NEEDED) return;  // workspace too small; cannot run safely
  const float* x      = (const float*)d_in[0];
  const float* eps_z  = (const float*)d_in[1];
  const float* eps_x  = (const float*)d_in[2];
  const float* W_x2z  = (const float*)d_in[3];
  const float* b_x2z  = (const float*)d_in[4];
  const float* W_xh2z = (const float*)d_in[5];
  const float* b_xh2z = (const float*)d_in[6];
  const float* W_z2x  = (const float*)d_in[7];
  const float* b_z2x  = (const float*)d_in[8];
  const float* W_zh2x = (const float*)d_in[9];
  const float* b_zh2x = (const float*)d_in[10];
  const float* W_dec  = (const float*)d_in[11];
  const float* b_dec  = (const float*)d_in[12];
  const float* W_prior= (const float*)d_in[13];
  const float* b_prior= (const float*)d_in[14];
  const float* W_ih   = (const float*)d_in[15];
  const float* b_ih   = (const float*)d_in[16];
  const float* W_hh   = (const float*)d_in[17];
  const float* b_hh   = (const float*)d_in[18];
  char* ws = (char*)d_ws;

  sisvae_prep<<<512, 256, 0, stream>>>(W_x2z, W_xh2z, W_z2x, W_zh2x, W_dec,
                                       W_prior, W_ih, W_hh, x, ws);
  sisvae_persistent<<<NBLK, TPB, 0, stream>>>(
      x, eps_z, eps_x, b_x2z, b_xh2z, b_z2x, b_zh2x, b_dec, b_prior, b_ih, b_hh,
      ws, (float*)d_out);
}
